// VectorQuantizer_76433237999928
// MI455X (gfx1250) — compile-verified, matrix operand order checked by reading, not run on hardware
//
#include <hip/hip_runtime.h>
#include <hip/hip_bf16.h>

#define NUM_CODES 1024
#define DIM 256
#define BATCH 32
#define TSZ 2048
#define NTOK (BATCH * TSZ)             // 65536 tokens
#define NQ ((size_t)BATCH * DIM * TSZ) // 16777216 quantized floats

typedef __attribute__((ext_vector_type(16))) __bf16        v16bf;
typedef __attribute__((ext_vector_type(8)))  float         v8f;
typedef __attribute__((ext_vector_type(8)))  unsigned int  v8u;
typedef __attribute__((ext_vector_type(4)))  unsigned int  u32x4;
typedef __attribute__((ext_vector_type(8)))  int           i32x8;
typedef __attribute__((ext_vector_type(4)))  int           i32x4;

__device__ __forceinline__ unsigned short f32_to_bf16(float f) {
  unsigned int u = __builtin_bit_cast(unsigned int, f);
  unsigned int r = u + 0x7FFFu + ((u >> 16) & 1u); // round-to-nearest-even
  return (unsigned short)(r >> 16);
}

// ---------------------------------------------------------------------------
// Kernel 1: ||e||^2 per codebook row. One wave (32 lanes) per code.
// ---------------------------------------------------------------------------
__global__ __launch_bounds__(256) void vq_enorm(const float* __restrict__ w,
                                                float* __restrict__ enorm) {
  int gwave = (blockIdx.x * blockDim.x + threadIdx.x) >> 5;
  int lane  = threadIdx.x & 31;
  if (gwave >= NUM_CODES) return;
  const float* row = w + (size_t)gwave * DIM;
  float s = 0.f;
#pragma unroll
  for (int i = 0; i < DIM / 32; ++i) {
    float v = row[lane + i * 32];
    s += v * v;
  }
#pragma unroll
  for (int m = 16; m >= 1; m >>= 1) s += __shfl_xor(s, m, 32);
  if (lane == 0) enorm[gwave] = s;
}

// ---------------------------------------------------------------------------
// Kernel 2: pre-swizzle codebook into per-lane WMMA B-fragments (bf16).
// Layout: bpack[((ct*8 + kc)*32 + lane) * 8 uints]  -> each code tile ct is a
// contiguous 8 KiB block (TDM-friendly).
// ---------------------------------------------------------------------------
__global__ __launch_bounds__(256) void vq_pack(const float* __restrict__ w,
                                               unsigned int* __restrict__ bpack) {
  int tid = blockIdx.x * blockDim.x + threadIdx.x;
  if (tid >= 64 * 8 * 32) return;
  int lane = tid & 31;
  int kc   = (tid >> 5) & 7;
  int ct   = tid >> 8;
  int ncol = lane & 15;
  int hi   = lane >> 4;
  int code = ct * 16 + ncol;
  const float* row = w + (size_t)code * DIM;
  unsigned int* dst = bpack + (size_t)tid * 8;
#pragma unroll
  for (int j = 0; j < 8; ++j) {
    int kbase = kc * 32 + ((j < 4) ? (hi * 8 + 2 * j) : (16 + hi * 8 + 2 * (j - 4)));
    unsigned int lo = f32_to_bf16(row[kbase]);
    unsigned int hb = f32_to_bf16(row[kbase + 1]);
    dst[j] = lo | (hb << 16);
  }
}

// ---------------------------------------------------------------------------
// Kernel 3: distance GEMM + argmin. One wave per 16-token tile; the block's
// 8 waves SHARE each 8 KiB code tile via LDS, double-buffered with the
// Tensor Data Mover (tensor_load_to_lds, TENSORcnt).
// dist(n, k) = ||e_k||^2 - 2 * x_n . e_k   (||x||^2 constant per row, dropped)
// ---------------------------------------------------------------------------
__global__ __launch_bounds__(256) void vq_argmin(const float* __restrict__ x,
                                                 const unsigned int* __restrict__ bpack,
                                                 const float* __restrict__ enorm,
                                                 int* __restrict__ idxOut,
                                                 float* __restrict__ idxOutF) {
  __shared__ unsigned char lbuf[2][8192];

  int tid   = threadIdx.x;
  int gwave = (blockIdx.x * blockDim.x + tid) >> 5; // tile id, 0..4095
  int lane  = tid & 31;
  int ncol  = lane & 15;
  int hi    = lane >> 4;
  int n0    = gwave * 16;

  // Never-executed store: keeps LLVM from proving lbuf store-free (TDM writes
  // it behind the compiler's back).
  if (blockIdx.x == 0x7fffffffu) lbuf[0][tid] = (unsigned char)tid;

  // ---- Preload A fragments: this lane carries row M = ncol, K subset by hi.
  v16bf afrag[8];
  {
    int n = n0 + ncol;            // tiles never straddle b: 2048 % 16 == 0
    int b = n >> 11;
    int t = n & 2047;
    const float* base = x + (size_t)b * DIM * TSZ + t; // element d at base[d*TSZ]
#pragma unroll
    for (int kc = 0; kc < 8; ++kc) {
      v8u pk;
#pragma unroll
      for (int j = 0; j < 8; ++j) {
        int kbase = kc * 32 + ((j < 4) ? (hi * 8 + 2 * j) : (16 + hi * 8 + 2 * (j - 4)));
        unsigned int lo = f32_to_bf16(base[(size_t)kbase * TSZ]);
        unsigned int hb = f32_to_bf16(base[(size_t)(kbase + 1) * TSZ]);
        pk[j] = lo | (hb << 16);
      }
      afrag[kc] = __builtin_bit_cast(v16bf, pk);
    }
  }

  float bestD[8];
  int   bestI[8];
#pragma unroll
  for (int v = 0; v < 8; ++v) { bestD[v] = 3.4e38f; bestI[v] = 0; }

  // ---- TDM setup: D# group1 (ISA §8.4): data_size=3 (8B), tensor_dim0=1024,
  // tensor_dim1=1, tile_dim0=1024, tensor_dim0_stride=1024, dim1_stride=1024.
  const bool issuer = (tid < 32); // wave 0 issues TDM for the whole block
  unsigned lbase = (unsigned)(uintptr_t)&lbuf[0][0];          // LDS byte addr
  unsigned long long gbase = (unsigned long long)(uintptr_t)bpack;
  i32x8 g1  = {0x00030000, 0, 0x04000000, 0x04010000, 0, 1024, 0x04000000, 0};
  i32x4 gz4 = {0, 0, 0, 0};
  i32x8 gz8 = {0, 0, 0, 0, 0, 0, 0, 0};

#define VQ_TDM_ISSUE(tile)                                                    \
  do {                                                                        \
    unsigned long long ga = gbase + (unsigned long long)(tile) * 8192ull;     \
    u32x4 g0;                                                                 \
    g0[0] = 1u;                             /* count=1, user mode */          \
    g0[1] = lbase + (unsigned)(((tile) & 1) * 8192); /* lds_addr */           \
    g0[2] = (unsigned)ga;                   /* global_addr[31:0] */           \
    g0[3] = (unsigned)(ga >> 32) | (2u << 30); /* addr[56:32] | type=2 */     \
    __builtin_amdgcn_tensor_load_to_lds(g0, g1, gz4, gz4, gz8, 0);            \
  } while (0)

  if (issuer) {
    VQ_TDM_ISSUE(0);
    __builtin_amdgcn_s_wait_tensorcnt(0);
  }
  __syncthreads();

  // ---- Sweep 64 code tiles of 16 codes; B tiles come from LDS.
  for (int ct = 0; ct < 64; ++ct) {
    if (issuer && (ct + 1) < 64) VQ_TDM_ISSUE(ct + 1); // prefetch next tile

    const v8u* bfp = (const v8u*)&lbuf[ct & 1][0];
    v8f acc = {};
#pragma unroll
    for (int kc = 0; kc < 8; ++kc) {
      v16bf bfrag = __builtin_bit_cast(v16bf, bfp[kc * 32 + lane]);
      acc = __builtin_amdgcn_wmma_f32_16x16x32_bf16(
          /*neg_a=*/false, afrag[kc], /*neg_b=*/false, bfrag,
          /*c_mod=*/(short)0, acc, /*reuse_a=*/false, /*reuse_b=*/false);
    }
    int   code = ct * 16 + ncol;   // column this lane owns in C
    float en   = enorm[code];
#pragma unroll
    for (int v = 0; v < 8; ++v) {  // C slot v -> row M = v + 8*hi
      float d = en - 2.0f * acc[v];
      if (d < bestD[v]) { bestD[v] = d; bestI[v] = code; }
    }

    if (issuer) __builtin_amdgcn_s_wait_tensorcnt(0); // next tile landed
    __syncthreads();  // all reads of cur done; next buffer ready for all
  }
#undef VQ_TDM_ISSUE

  // ---- Butterfly (dist, idx) min within each 16-lane group; tie -> lower idx.
#pragma unroll
  for (int v = 0; v < 8; ++v) {
    float d = bestD[v];
    int   i = bestI[v];
#pragma unroll
    for (int m = 8; m >= 1; m >>= 1) {
      float od = __shfl_xor(d, m, 32);
      int   oi = __shfl_xor(i, m, 32);
      if (od < d || (od == d && oi < i)) { d = od; i = oi; }
    }
    if (ncol == 0) {
      int n = n0 + v + 8 * hi;
      idxOut[n]  = i;
      idxOutF[n] = (float)i;
    }
  }
}

// ---------------------------------------------------------------------------
// Kernel 4: gather winning codewords back into [B, D, T] layout.
// Thread per token; stores coalesced across threads (consecutive t).
// ---------------------------------------------------------------------------
__global__ __launch_bounds__(256) void vq_gather(const float* __restrict__ w,
                                                 const int* __restrict__ idx,
                                                 float* __restrict__ out) {
  int n = blockIdx.x * blockDim.x + threadIdx.x;
  if (n >= NTOK) return;
  int b = n >> 11;
  int t = n & 2047;
  const float* wr = w + (size_t)idx[n] * DIM;
  float* o = out + (size_t)b * DIM * TSZ + t;
#pragma unroll 4
  for (int d = 0; d < DIM; ++d) o[(size_t)d * TSZ] = wr[d];
}

// ---------------------------------------------------------------------------
extern "C" void kernel_launch(void* const* d_in, const int* in_sizes, int n_in,
                              void* d_out, int out_size, void* d_ws, size_t ws_size,
                              hipStream_t stream) {
  const float* x = (const float*)d_in[0]; // [32, 256, 2048]
  const float* w = (const float*)d_in[1]; // [1024, 256]
  float* out = (float*)d_out;             // quantized (NQ) then indices (NTOK)

  unsigned char* ws = (unsigned char*)d_ws;
  unsigned int* bpack = (unsigned int*)ws;                  // 512 KiB
  float*        enorm = (float*)(ws + 512 * 1024);          //   4 KiB
  int*          idxA  = (int*)(ws + 512 * 1024 + 4 * 1024); // 256 KiB

  vq_enorm <<<128, 256, 0, stream>>>(w, enorm);
  vq_pack  <<< 64, 256, 0, stream>>>(w, bpack);
  vq_argmin<<<512, 256, 0, stream>>>(x, bpack, enorm, idxA, out + NQ);
  vq_gather<<<256, 256, 0, stream>>>(w, idxA, out);
}
